// PerformerBlock_47519518163145
// MI455X (gfx1250) — matrix operand order repaired and checked by reading, not compiled, not run on hardware
//
#include <hip/hip_runtime.h>
#include <hip/hip_bf16.h>
#include <math.h>

// ---------------------------------------------------------------------------
// Performer block on gfx1250: bf16 WMMA GEMMs (v_wmma_f32_16x16x32_bf16)
// with fp32 accumulation; small VALU kernels for LN / kvs / den.
// All WMMA paths are wave-uniform straight-line code (EXEC all-1s).
// GEMM inner loop is explicitly software-pipelined (double-buffered
// fragment sets rotated across a 2x-unrolled steady state).
// ---------------------------------------------------------------------------

typedef __bf16 bf16_t;
typedef __bf16 v16bf __attribute__((ext_vector_type(16)));
typedef __bf16 v8bf  __attribute__((ext_vector_type(8)));
typedef float  v8f   __attribute__((ext_vector_type(8)));

struct bfrag2 { v8bf lo, hi; };

static __device__ __forceinline__ float  b2f(bf16_t x) { return (float)x; }
static __device__ __forceinline__ bf16_t f2b(float x)  { return (bf16_t)x; }

static __device__ __forceinline__ float gelu_exact(float x) {
    return 0.5f * x * (1.0f + erff(x * 0.70710678118654752f));
}

// ---------------------------------------------------------------------------
// Weight convert + transpose: in fp32 [K,N] row-major -> out bf16 [N,K]
// ---------------------------------------------------------------------------
__global__ __launch_bounds__(256)
void transpose_convert_kernel(const float* __restrict__ in, int K, int N,
                              bf16_t* __restrict__ out) {
    __shared__ float tile[32][33];
    int tx = threadIdx.x;           // 0..31
    int ty = threadIdx.y;           // 0..7
    int n0 = blockIdx.x * 32;
    int k0 = blockIdx.y * 32;
    #pragma unroll
    for (int i = 0; i < 4; ++i) {
        int k = k0 + ty + i * 8;
        int n = n0 + tx;
        if (k < K && n < N) tile[ty + i * 8][tx] = in[(size_t)k * N + n];
    }
    __syncthreads();
    #pragma unroll
    for (int i = 0; i < 4; ++i) {
        int n = n0 + ty + i * 8;
        int k = k0 + tx;
        if (k < K && n < N) out[(size_t)n * K + k] = f2b(tile[tx][ty + i * 8]);
    }
}

// ---------------------------------------------------------------------------
// LayerNorm over E=1024 (4 elems/thread, 256 threads/token) -> bf16
// ---------------------------------------------------------------------------
__global__ __launch_bounds__(256)
void ln_kernel(const float* __restrict__ x, const float* __restrict__ gamma,
               const float* __restrict__ beta, bf16_t* __restrict__ y, int E) {
    size_t tok = blockIdx.x;
    int t = threadIdx.x;
    const float* xr = x + tok * (size_t)E;
    float vals[4];
    float s = 0.f, s2 = 0.f;
    #pragma unroll
    for (int i = 0; i < 4; ++i) {
        float v = xr[t + i * 256];
        vals[i] = v; s += v; s2 += v * v;
    }
    __shared__ float r1[256], r2[256];
    r1[t] = s; r2[t] = s2;
    __syncthreads();
    for (int o = 128; o > 0; o >>= 1) {
        if (t < o) { r1[t] += r1[t + o]; r2[t] += r2[t + o]; }
        __syncthreads();
    }
    float mu   = r1[0] / (float)E;
    float var  = r2[0] / (float)E - mu * mu;
    float rstd = rsqrtf(var + 1e-6f);
    #pragma unroll
    for (int i = 0; i < 4; ++i) {
        int c = t + i * 256;
        y[tok * (size_t)E + c] = f2b((vals[i] - mu) * rstd * gamma[c] + beta[c]);
    }
}

// ---------------------------------------------------------------------------
// kvs[m,d] = sum_l kp[l,m]*v[l,d]  and  ksum[m] = sum_l kp[l,m], per (b,h).
// Writes kvsT[z][d][m] (bf16) so `num` runs through the WMMA GEMM.
// ---------------------------------------------------------------------------
__global__ __launch_bounds__(256)
void kvs_kernel(const bf16_t* __restrict__ kp, const bf16_t* __restrict__ v,
                bf16_t* __restrict__ kvsT, float* __restrict__ ksum,
                int Hn, int L, int E) {
    int z = blockIdx.x;
    int b = z / Hn, h = z % Hn;
    int t = threadIdx.x;
    int d  = t & 63;
    int mg = t >> 6;                 // m-group 0..3 -> m = mg*16 + j
    const bf16_t* kpb = kp + (size_t)b * L * E + h * 64;
    const bf16_t* vb  = v  + (size_t)b * L * E + h * 64;
    float acc[16], ks[16];
    #pragma unroll
    for (int j = 0; j < 16; ++j) { acc[j] = 0.f; ks[j] = 0.f; }
    for (int l = 0; l < L; ++l) {
        size_t off = (size_t)l * E;
        float vv = b2f(vb[off + d]);
        #pragma unroll
        for (int j = 0; j < 16; ++j) {
            float kk = b2f(kpb[off + mg * 16 + j]);
            acc[j] += kk * vv;
            ks[j]  += kk;
        }
    }
    bf16_t* o = kvsT + (size_t)z * 64 * 64 + (size_t)d * 64 + mg * 16;
    #pragma unroll
    for (int j = 0; j < 16; ++j) o[j] = f2b(acc[j]);
    if (d == 0) {
        float* ko = ksum + (size_t)z * 64 + mg * 16;
        #pragma unroll
        for (int j = 0; j < 16; ++j) ko[j] = ks[j];
    }
}

// ---------------------------------------------------------------------------
// den[b,l,h] = dot(qp[b,l,h,:], ksum[b,h,:]);  att = num/den -> bf16
// ---------------------------------------------------------------------------
__global__ __launch_bounds__(256)
void att_kernel(const bf16_t* __restrict__ qp, const float* __restrict__ ksum,
                const float* __restrict__ num, bf16_t* __restrict__ att,
                int Hn, int L, int E) {
    size_t tok = blockIdx.x;
    int b = (int)(tok / (size_t)L);
    int t = threadIdx.x;
    __shared__ float sden[16];
    if (t < 16) sden[t] = 0.f;
    __syncthreads();
    int h = t >> 4, i = t & 15;
    const bf16_t* q  = qp + tok * (size_t)E + h * 64;
    const float*  ks = ksum + ((size_t)b * Hn + h) * 64;
    float p = 0.f;
    #pragma unroll
    for (int j = 0; j < 4; ++j) {
        int m = i * 4 + j;
        p += b2f(q[m]) * ks[m];
    }
    atomicAdd(&sden[h], p);
    __syncthreads();
    for (int c = t; c < E; c += 256) {
        att[tok * (size_t)E + c] = f2b(num[tok * (size_t)E + c] / sden[c >> 6]);
    }
}

// ---------------------------------------------------------------------------
// Fragment set for one K-step (32): A fragment + 4 B fragments.
// ---------------------------------------------------------------------------
struct FragSet {
    v16bf a;
    v16bf b[4];
};

static __device__ __forceinline__
void load_frags(FragSet& f, const bf16_t* __restrict__ aRow,
                const bf16_t* __restrict__ bRow, size_t bStep,
                int kb, int koffA) {
    // A fragment (16x32 bf16): lane<16 holds K {0..15}, lane>=16 {16..31}
    bfrag2 af;
    af.lo = *(const v8bf*)(aRow + kb + koffA);
    af.hi = *(const v8bf*)(aRow + kb + 8 + koffA);
    f.a = __builtin_bit_cast(v16bf, af);
    // B fragments (32x16 bf16): lane%16 = N-col, 16 contiguous K per half
    #pragma unroll
    for (int t = 0; t < 4; ++t) {
        f.b[t] = *(const v16bf*)(bRow + (size_t)t * bStep + kb);
    }
}

// ---------------------------------------------------------------------------
// bf16 WMMA GEMM:  C[M,N] = A[M,K] * Bt[N,K]^T, fp32 accumulate.
// Block = 256 threads = 8 waves arranged WAVES_M x WAVES_N (product == 8).
// Each wave owns a 16(M) x 64(N) strip. The grid tiles M and N EXACTLY
// (no runtime guards) so the hot path is wave-uniform straight-line code and
// EXEC stays all-1s around every WMMA (ISA 7.12 requirement).
// Software pipelining: two FragSets rotated across a 2x-unrolled loop, so
// each 4-WMMA burst overlaps the next K-step's 10 in-flight b128 loads.
// Requires K/32 even (true for K in {1024,4096,64}); the final rotated load
// overreads 64B/row into adjacent workspace regions (mapped, never read back).
// Batched over grid.z with (b,h) stride decomposition: z = b*HH + h.
// MODE: 0 plain->bf16 | 1 relu+1e-3->bf16 | 2 bias+gelu->bf16
//       3 bias+gelu+resid->f32 | 4 resid->f32 | 5 plain->f32
// A/B use contiguous K halves (k = 16*half + s) in both; C/D per CDNA5 ISA 7.12.2 (wave32).
// ---------------------------------------------------------------------------
template <int MODE, int WAVES_M, int WAVES_N>
__global__ __launch_bounds__(256)
void gemm_bf16_kernel(const bf16_t* __restrict__ A, int lda,
                      const bf16_t* __restrict__ Bt, int ldb,
                      void* __restrict__ Cv, int ldc,
                      const float* __restrict__ bias,
                      const float* __restrict__ resid,
                      int K,
                      int HH,
                      long long sAb, long long sAh,
                      long long sBb, long long sBh,
                      long long sCb, long long sCh) {
    int z = blockIdx.z;
    int b = z / HH, h = z % HH;
    A  += (size_t)((long long)b * sAb + (long long)h * sAh);
    Bt += (size_t)((long long)b * sBb + (long long)h * sBh);
    size_t cOff = (size_t)((long long)b * sCb + (long long)h * sCh);

    int lane = threadIdx.x & 31;
    int w    = threadIdx.x >> 5;
    int wy   = w % WAVES_M;
    int wx   = w / WAVES_M;
    int m0 = blockIdx.y * (WAVES_M * 16) + wy * 16;
    int n0 = blockIdx.x * (WAVES_N * 64) + wx * 64;

    int half  = lane >> 4;          // 0 or 1
    int koffA = half ? 16 : 0;      // A per-lane K offset (contiguous halves)
    int koffB = half ? 16 : 0;      // B per-lane K offset (contiguous halves)
    int row   = m0 + (lane & 15);
    const bf16_t* aRow = A + (size_t)row * lda;
    const bf16_t* bRow = Bt + (size_t)(n0 + (lane & 15)) * ldb + koffB;
    size_t bStep = (size_t)16 * ldb;   // next 16-column subtile

    v8f acc[4] = {};

    FragSet f0, f1;
    load_frags(f0, aRow, bRow, bStep, 0, koffA);

    // K/32 is even for every call site; process K-steps in pairs with
    // rotated double buffering (no cross-buffer copies).
    for (int kb = 0; kb < K; kb += 64) {
        __builtin_prefetch(aRow + kb + 512, 0, 1);
        load_frags(f1, aRow, bRow, bStep, kb + 32, koffA);
        #pragma unroll
        for (int t = 0; t < 4; ++t) {
            acc[t] = __builtin_amdgcn_wmma_f32_16x16x32_bf16(
                false, f0.a, false, f0.b[t], (short)0, acc[t], false, false);
        }
        load_frags(f0, aRow, bRow, bStep, kb + 64, koffA); // tail: benign overread
        #pragma unroll
        for (int t = 0; t < 4; ++t) {
            acc[t] = __builtin_amdgcn_wmma_f32_16x16x32_bf16(
                false, f1.a, false, f1.b[t], (short)0, acc[t], false, false);
        }
    }

    // C/D layout: lane%16 = N-col; VGPR r = M-row (r, lane>=16 adds +8)
    int colBase = lane & 15;
    int rbase   = m0 + (half ? 8 : 0);
    #pragma unroll
    for (int t = 0; t < 4; ++t) {
        int col = n0 + t * 16 + colBase;
        #pragma unroll
        for (int r = 0; r < 8; ++r) {
            int rr = rbase + r;
            float v = acc[t][r];
            size_t idx = (size_t)rr * ldc + col + cOff;
            if (MODE == 0) {
                ((bf16_t*)Cv)[idx] = f2b(v);
            } else if (MODE == 1) {
                ((bf16_t*)Cv)[idx] = f2b(fmaxf(v, 0.f) + 1e-3f);
            } else if (MODE == 2) {
                ((bf16_t*)Cv)[idx] = f2b(gelu_exact(v + bias[col]));
            } else if (MODE == 3) {
                ((float*)Cv)[idx] = gelu_exact(v + bias[col]) + resid[idx];
            } else if (MODE == 4) {
                ((float*)Cv)[idx] = v + resid[idx];
            } else { // 5
                ((float*)Cv)[idx] = v;
            }
        }
    }
}

// ---------------------------------------------------------------------------
// Host launcher
// ---------------------------------------------------------------------------
extern "C" void kernel_launch(void* const* d_in, const int* in_sizes, int n_in,
                              void* d_out, int out_size, void* d_ws, size_t ws_size,
                              hipStream_t stream) {
    (void)in_sizes; (void)n_in; (void)out_size; (void)ws_size;
    const int Bn = 8, L = 2048, E = 1024, H = 16, dh = 64, F = 4096;
    const int M = Bn * L; // 16384 tokens

    const float* x      = (const float*)d_in[0];
    const float* gamma1 = (const float*)d_in[1];
    const float* beta1  = (const float*)d_in[2];
    const float* wq     = (const float*)d_in[3];
    const float* wk     = (const float*)d_in[4];
    const float* wv     = (const float*)d_in[5];
    const float* wo     = (const float*)d_in[6];
    const float* w1     = (const float*)d_in[7];
    const float* b1     = (const float*)d_in[8];
    const float* w2     = (const float*)d_in[9];
    const float* b2     = (const float*)d_in[10];
    float* out = (float*)d_out;

    char* ws = (char*)d_ws;
    const size_t MB = 1ull << 20;
    bf16_t* wqT  = (bf16_t*)(ws + 0 * MB);     // [1024,1024] bf16 (2MB)
    bf16_t* wkT  = (bf16_t*)(ws + 2 * MB);
    bf16_t* wvT  = (bf16_t*)(ws + 4 * MB);
    bf16_t* woT  = (bf16_t*)(ws + 6 * MB);
    bf16_t* w1T  = (bf16_t*)(ws + 8 * MB);     // [4096,1024] (8MB)
    bf16_t* w2T  = (bf16_t*)(ws + 16 * MB);    // [1024,4096] (8MB)
    bf16_t* lnb  = (bf16_t*)(ws + 24 * MB);    // [16384,1024] (32MB) ln1 then ln2
    bf16_t* qpb  = (bf16_t*)(ws + 56 * MB);    // qp (32MB)
    bf16_t* kpb  = (bf16_t*)(ws + 88 * MB);    // kp, later reused as att (32MB)
    bf16_t* vb   = (bf16_t*)(ws + 120 * MB);   // v (32MB)
    float*  num  = (float*)(ws + 152 * MB);    // [16384,1024] f32 (64MB)
    bf16_t* hb   = (bf16_t*)(ws + 56 * MB);    // [16384,4096] (128MB) reuses qp..num
    float*  add1 = (float*)(ws + 216 * MB);    // [16384,1024] f32 (64MB)
    bf16_t* kvsT = (bf16_t*)(ws + 280 * MB);   // [128,64,64] bf16 (1MB)
    float*  ksum = (float*)(ws + 281 * MB);    // [128,64] f32

    dim3 blk(256);
    dim3 tblk(32, 8);

    // --- weight convert + transpose to bf16 [N,K] ---
    transpose_convert_kernel<<<dim3(32, 32),  tblk, 0, stream>>>(wq, E, E, wqT);
    transpose_convert_kernel<<<dim3(32, 32),  tblk, 0, stream>>>(wk, E, E, wkT);
    transpose_convert_kernel<<<dim3(32, 32),  tblk, 0, stream>>>(wv, E, E, wvT);
    transpose_convert_kernel<<<dim3(32, 32),  tblk, 0, stream>>>(wo, E, E, woT);
    transpose_convert_kernel<<<dim3(128, 32), tblk, 0, stream>>>(w1, E, F, w1T);
    transpose_convert_kernel<<<dim3(32, 128), tblk, 0, stream>>>(w2, F, E, w2T);

    // --- LN1 ---
    ln_kernel<<<M, blk, 0, stream>>>(x, gamma1, beta1, lnb, E);

    // --- q/k/v projections (WMMA), epilogue relu+STAB for q,k ---
    dim3 gP(E / 128, M / 64, 1); // (8,256)
    gemm_bf16_kernel<1, 4, 2><<<gP, blk, 0, stream>>>(lnb, E, wqT, E, (void*)qpb, E,
        nullptr, nullptr, E, 1, 0, 0, 0, 0, 0, 0);
    gemm_bf16_kernel<1, 4, 2><<<gP, blk, 0, stream>>>(lnb, E, wkT, E, (void*)kpb, E,
        nullptr, nullptr, E, 1, 0, 0, 0, 0, 0, 0);
    gemm_bf16_kernel<0, 4, 2><<<gP, blk, 0, stream>>>(lnb, E, wvT, E, (void*)vb, E,
        nullptr, nullptr, E, 1, 0, 0, 0, 0, 0, 0);

    // --- kvs + ksum per (b,h) ---
    kvs_kernel<<<Bn * H, blk, 0, stream>>>(kpb, vb, kvsT, ksum, H, L, E);

    // --- num = qp @ kvs (batched WMMA over z = b*H + h); N=64 exact with
    //     8x1 wave arrangement (one wave covers all 64 columns) ---
    dim3 gN(1, L / 128, Bn * H); // (1,16,128)
    gemm_bf16_kernel<5, 8, 1><<<gN, blk, 0, stream>>>(
        qpb, E, kvsT, dh, (void*)num, E, nullptr, nullptr,
        dh, H,
        (long long)L * E, (long long)dh,            // A: b*L*E + h*64
        (long long)H * dh * dh, (long long)dh * dh, // B: z*4096
        (long long)L * E, (long long)dh);           // C: b*L*E + h*64

    // --- den + normalize -> att (bf16, overwrites kp buffer) ---
    att_kernel<<<M, blk, 0, stream>>>(qpb, ksum, num, kpb, H, L, E);

    // --- attn_out @ wo + x -> add1 (f32) ---
    gemm_bf16_kernel<4, 4, 2><<<gP, blk, 0, stream>>>(kpb, E, woT, E, (void*)add1, E,
        nullptr, x, E, 1, 0, 0, 0, 0, 0, 0);

    // --- LN2 (reference reuses gamma1/beta1) ---
    ln_kernel<<<M, blk, 0, stream>>>(add1, gamma1, beta1, lnb, E);

    // --- FFN1: gelu(ln2 @ w1 + b1) -> h (bf16) ---
    dim3 gF1(F / 128, M / 64, 1); // (32,256)
    gemm_bf16_kernel<2, 4, 2><<<gF1, blk, 0, stream>>>(lnb, E, w1T, E, (void*)hb, F,
        b1, nullptr, E, 1, 0, 0, 0, 0, 0, 0);

    // --- FFN2: gelu(h @ w2 + b2) + add1 -> out (f32) ---
    dim3 gF2(E / 128, M / 64, 1); // (8,256)
    gemm_bf16_kernel<3, 4, 2><<<gF2, blk, 0, stream>>>(hb, F, w2T, F, (void*)out, E,
        b2, add1, E, 1, 0, 0, 0, 0, 0, 0);
}
